// DMPNN_83640193122798
// MI455X (gfx1250) — matrix operand order repaired and checked by reading, not compiled, not run on hardware
//
#include <hip/hip_runtime.h>

// ---------------------------------------------------------------------------
// CDNA5 (gfx1250) D-MPNN: bf16 WMMA GEMMs + fp32 scatter/segment reductions.
// ---------------------------------------------------------------------------

typedef __attribute__((ext_vector_type(16))) __bf16        v16bf;
typedef __attribute__((ext_vector_type(8)))  float         v8f;
typedef __attribute__((ext_vector_type(4)))  unsigned int  u32x4;

static constexpr int kAtoms  = 200000;
static constexpr int kBonds  = 400000;
static constexpr int kMols   = 10000;
static constexpr int kAF     = 133;
static constexpr int kBF     = 13;
static constexpr int kHid    = 300;
static constexpr int kTasks  = 12;
static constexpr int kDepth  = 3;

// K padded to multiples of 32 (one bf16 WMMA step), N padded to 320 (5 groups of 64)
static constexpr int kKi     = 160;   // pad of 146
static constexpr int kKh     = 320;   // pad of 300
static constexpr int kKo     = 448;   // pad of 433
static constexpr int kNPad   = 320;
static constexpr int kNGrp   = kNPad / 64;   // 5 groups of 4 n-tiles per wave

union Frag { v16bf v; u32x4 u[2]; };

// ---------------------------------------------------------------------------
// WMMA GEMM: C[M, kNPad] = A[M, Kpad](bf16) * W[kNPad, Kpad](bf16)^T
// One wave -> 16 rows x 64 cols (4 accumulators sharing one A fragment).
// EPI 0: h0 = h = relu(C)            (init bond messages)
// EPI 1: h  = relu(h0 + C)           (message update, fused residual)
// EPI 2: atomicAdd mol_vecs[mol_ids] += relu(C + b_o)   (fused readout pool)
// ---------------------------------------------------------------------------
template<int EPI>
__launch_bounds__(256)
__global__ void gemm_wmma_bf16(const __bf16* __restrict__ A, int lda,
                               const __bf16* __restrict__ Bw, int ldb,
                               int Mtiles, int Ktiles,
                               const __bf16* __restrict__ h0,
                               __bf16* __restrict__ hOut,
                               __bf16* __restrict__ h0Out,
                               const float* __restrict__ bias,
                               const int* __restrict__ molIds,
                               float* __restrict__ molVecs)
{
    const int wid = (int)((blockIdx.x * blockDim.x + threadIdx.x) >> 5);
    if (wid >= Mtiles * kNGrp) return;          // uniform per wave -> EXEC all-ones
    const int lane  = threadIdx.x & 31;
    const int mt    = wid / kNGrp;
    const int ng    = wid - mt * kNGrp;
    const int khalf = lane >> 4;                // which K-half this lane holds
    const int l15   = lane & 15;

    const __bf16* arow = A  + (size_t)(mt * 16 + l15) * lda;
    const __bf16* brow = Bw + (size_t)(ng * 64 + l15) * ldb;

    v8f acc0 = {}, acc1 = {}, acc2 = {}, acc3 = {};
    for (int kt = 0; kt < Ktiles; ++kt) {
        const int k0 = kt * 32;
        if (kt + 1 < Ktiles) __builtin_prefetch(arow + k0 + 32, 0, 3);

        // A 16x32 bf16 fragment: lane holds row (lane&15); elems 0..7 -> K=k0+khalf*8+i,
        // elems 8..15 -> K=k0+16+khalf*8+(i-8). Two contiguous 16B loads.
        Frag a;
        a.u[0] = *(const u32x4*)(arow + k0 + khalf * 8);
        a.u[1] = *(const u32x4*)(arow + k0 + 16 + khalf * 8);

        // B 32x16 bf16 fragment: lane holds col (lane&15); same K order as A.
        const int boff = k0 + khalf * 8;
        Frag b0, b1, b2, b3;
        b0.u[0] = *(const u32x4*)(brow            + boff);
        b0.u[1] = *(const u32x4*)(brow            + boff + 16);
        b1.u[0] = *(const u32x4*)(brow + 16 * ldb + boff);
        b1.u[1] = *(const u32x4*)(brow + 16 * ldb + boff + 16);
        b2.u[0] = *(const u32x4*)(brow + 32 * ldb + boff);
        b2.u[1] = *(const u32x4*)(brow + 32 * ldb + boff + 16);
        b3.u[0] = *(const u32x4*)(brow + 48 * ldb + boff);
        b3.u[1] = *(const u32x4*)(brow + 48 * ldb + boff + 16);

        acc0 = __builtin_amdgcn_wmma_f32_16x16x32_bf16(false, a.v, false, b0.v, (short)0, acc0, false, false);
        acc1 = __builtin_amdgcn_wmma_f32_16x16x32_bf16(false, a.v, false, b1.v, (short)0, acc1, false, false);
        acc2 = __builtin_amdgcn_wmma_f32_16x16x32_bf16(false, a.v, false, b2.v, (short)0, acc2, false, false);
        acc3 = __builtin_amdgcn_wmma_f32_16x16x32_bf16(false, a.v, false, b3.v, (short)0, acc3, false, false);
    }

    // C/D layout: lane (l15) = col, VGPR r = row r + 8*khalf within the 16x16 tile.
    const int mbase = mt * 16 + khalf * 8;
    #pragma unroll
    for (int t = 0; t < 4; ++t) {
        const v8f acc = (t == 0) ? acc0 : (t == 1) ? acc1 : (t == 2) ? acc2 : acc3;
        const int n = ng * 64 + t * 16 + l15;
        if (n < kHid) {
            #pragma unroll
            for (int r = 0; r < 8; ++r) {
                const int m = mbase + r;
                const float c = acc[r];
                if constexpr (EPI == 0) {
                    const __bf16 v = (__bf16)fmaxf(c, 0.0f);
                    h0Out[(size_t)m * kHid + n] = v;
                    hOut [(size_t)m * kHid + n] = v;
                } else if constexpr (EPI == 1) {
                    const float x = c + (float)h0[(size_t)m * kHid + n];
                    hOut[(size_t)m * kHid + n] = (__bf16)fmaxf(x, 0.0f);
                } else {
                    const float x = fmaxf(c + bias[n], 0.0f);
                    atomicAdd(&molVecs[(size_t)molIds[m] * kHid + n], x);
                }
            }
        }
    }
}

// ------------------------- elementwise / prep kernels -----------------------

__global__ void k_pad_weight(const float* __restrict__ src, __bf16* __restrict__ dst,
                             int rs, int cs, int cd, size_t n)
{
    size_t i = blockIdx.x * 256ull + threadIdx.x;
    if (i >= n) return;
    int r = (int)(i / cd);
    int c = (int)(i - (size_t)r * cd);
    float v = (r < rs && c < cs) ? src[(size_t)r * cs + c] : 0.0f;
    dst[i] = (__bf16)v;
}

__global__ void k_target(const int* __restrict__ b2a, const int* __restrict__ b2revb,
                         int* __restrict__ tgt)
{
    int i = blockIdx.x * 256 + threadIdx.x;
    if (i >= kBonds) return;
    tgt[i] = b2a[b2revb[i]];
}

// A_i = concat(f_atoms[b2a], f_bonds), bf16, K padded to kKi
__global__ void k_build_ai(const float* __restrict__ fa, const float* __restrict__ fb,
                           const int* __restrict__ b2a, __bf16* __restrict__ A)
{
    size_t i = blockIdx.x * 256ull + threadIdx.x;
    if (i >= (size_t)kBonds * kKi) return;
    int b = (int)(i / kKi);
    int j = (int)(i - (size_t)b * kKi);
    float v = 0.0f;
    if (j < kAF)             v = fa[(size_t)b2a[b] * kAF + j];
    else if (j < kAF + kBF)  v = fb[(size_t)b * kBF + (j - kAF)];
    A[i] = (__bf16)v;
}

__global__ void k_zero_f32(float* __restrict__ p, size_t n)
{
    size_t i = blockIdx.x * 256ull + threadIdx.x;
    if (i < n) p[i] = 0.0f;
}

// atom_msg[tgt[b]] += h[b]   (fp32 accumulation of bf16 messages)
__global__ void k_scatter(const __bf16* __restrict__ h, const int* __restrict__ tgt,
                          float* __restrict__ am)
{
    size_t i = blockIdx.x * 256ull + threadIdx.x;
    if (i >= (size_t)kBonds * kHid) return;
    int b = (int)(i / kHid);
    int j = (int)(i - (size_t)b * kHid);
    atomicAdd(&am[(size_t)tgt[b] * kHid + j], (float)h[i]);
}

// m = atom_msg[b2a] - h[b2revb], bf16, K padded to kKh
__global__ void k_build_m(const float* __restrict__ am, const __bf16* __restrict__ h,
                          const int* __restrict__ b2a, const int* __restrict__ b2revb,
                          __bf16* __restrict__ A)
{
    size_t i = blockIdx.x * 256ull + threadIdx.x;
    if (i >= (size_t)kBonds * kKh) return;
    int b = (int)(i / kKh);
    int j = (int)(i - (size_t)b * kKh);
    float v = 0.0f;
    if (j < kHid)
        v = am[(size_t)b2a[b] * kHid + j] - (float)h[(size_t)b2revb[b] * kHid + j];
    A[i] = (__bf16)v;
}

// A_o = concat(f_atoms, atom_msg), bf16, K padded to kKo
__global__ void k_build_ao(const float* __restrict__ fa, const float* __restrict__ am,
                           __bf16* __restrict__ A)
{
    size_t i = blockIdx.x * 256ull + threadIdx.x;
    if (i >= (size_t)kAtoms * kKo) return;
    int a = (int)(i / kKo);
    int j = (int)(i - (size_t)a * kKo);
    float v = 0.0f;
    if (j < kAF)              v = fa[(size_t)a * kAF + j];
    else if (j < kAF + kHid)  v = am[(size_t)a * kHid + (j - kAF)];
    A[i] = (__bf16)v;
}

// out = relu(mol_vecs @ W_r1^T + b_r1)   (fp32, ~1.8 GFLOP: negligible)
__global__ void k_r1(const float* __restrict__ mv, const float* __restrict__ W,
                     const float* __restrict__ b, float* __restrict__ out)
{
    int i = blockIdx.x * 256 + threadIdx.x;
    if (i >= kMols * kHid) return;
    int mol = i / kHid, c = i - mol * kHid;
    const float* mrow = mv + (size_t)mol * kHid;
    const float* wrow = W + (size_t)c * kHid;
    float s = b[c];
    for (int k = 0; k < kHid; ++k) s += mrow[k] * wrow[k];
    out[i] = fmaxf(s, 0.0f);
}

// logits = out @ W_r2^T + b_r2
__global__ void k_r2(const float* __restrict__ in, const float* __restrict__ W,
                     const float* __restrict__ b, float* __restrict__ logits)
{
    int i = blockIdx.x * 256 + threadIdx.x;
    if (i >= kMols * kTasks) return;
    int mol = i / kTasks, t = i - mol * kTasks;
    const float* irow = in + (size_t)mol * kHid;
    const float* wrow = W + (size_t)t * kHid;
    float s = b[t];
    for (int k = 0; k < kHid; ++k) s += irow[k] * wrow[k];
    logits[i] = s;
}

// ------------------------------ workspace map -------------------------------
static constexpr size_t au256(size_t x) { return (x + 255) & ~(size_t)255; }
static constexpr size_t OFF_TGT = 0;
static constexpr size_t OFF_H0  = au256(OFF_TGT + (size_t)kBonds * 4);
static constexpr size_t OFF_H   = au256(OFF_H0 + (size_t)kBonds * kHid * 2);
static constexpr size_t OFF_A   = au256(OFF_H  + (size_t)kBonds * kHid * 2);
static constexpr size_t A_ELEMS = (size_t)kBonds * kKh;   // largest A operand
static constexpr size_t OFF_AM  = au256(OFF_A  + A_ELEMS * 2);
static constexpr size_t OFF_MV  = au256(OFF_AM + (size_t)kAtoms * kHid * 4);
static constexpr size_t OFF_R1  = au256(OFF_MV + (size_t)kMols * kHid * 4);
static constexpr size_t OFF_WI  = au256(OFF_R1 + (size_t)kMols * kHid * 4);
static constexpr size_t OFF_WH  = au256(OFF_WI + (size_t)kNPad * kKi * 2);
static constexpr size_t OFF_WO  = au256(OFF_WH + (size_t)kNPad * kKh * 2);

static inline int nblk(size_t n) { return (int)((n + 255) / 256); }

extern "C" void kernel_launch(void* const* d_in, const int* in_sizes, int n_in,
                              void* d_out, int out_size, void* d_ws, size_t ws_size,
                              hipStream_t stream)
{
    (void)in_sizes; (void)n_in; (void)out_size; (void)ws_size;

    const float* f_atoms = (const float*)d_in[0];
    const float* f_bonds = (const float*)d_in[1];
    const int*   b2a     = (const int*)  d_in[2];
    const int*   b2revb  = (const int*)  d_in[3];
    const int*   mol_ids = (const int*)  d_in[4];
    const float* W_i     = (const float*)d_in[5];
    const float* W_h     = (const float*)d_in[6];
    const float* W_o     = (const float*)d_in[7];
    const float* b_o     = (const float*)d_in[8];
    const float* W_r1    = (const float*)d_in[9];
    const float* b_r1    = (const float*)d_in[10];
    const float* W_r2    = (const float*)d_in[11];
    const float* b_r2    = (const float*)d_in[12];
    float* logits = (float*)d_out;

    char* ws = (char*)d_ws;
    int*    tgt  = (int*)   (ws + OFF_TGT);
    __bf16* h0   = (__bf16*)(ws + OFF_H0);
    __bf16* h    = (__bf16*)(ws + OFF_H);
    __bf16* Abuf = (__bf16*)(ws + OFF_A);
    float*  am   = (float*) (ws + OFF_AM);
    float*  mv   = (float*) (ws + OFF_MV);
    float*  r1o  = (float*) (ws + OFF_R1);
    __bf16* WiP  = (__bf16*)(ws + OFF_WI);
    __bf16* WhP  = (__bf16*)(ws + OFF_WH);
    __bf16* WoP  = (__bf16*)(ws + OFF_WO);

    constexpr int MT_B = kBonds / 16;   // 25000
    constexpr int MT_A = kAtoms / 16;   // 12500

    // Pad + convert weights to bf16 (rebuilt every call: deterministic).
    k_pad_weight<<<nblk((size_t)kNPad * kKi), 256, 0, stream>>>(W_i, WiP, kHid, kAF + kBF, kKi, (size_t)kNPad * kKi);
    k_pad_weight<<<nblk((size_t)kNPad * kKh), 256, 0, stream>>>(W_h, WhP, kHid, kHid,      kKh, (size_t)kNPad * kKh);
    k_pad_weight<<<nblk((size_t)kNPad * kKo), 256, 0, stream>>>(W_o, WoP, kHid, kAF + kHid, kKo, (size_t)kNPad * kKo);
    k_target<<<nblk(kBonds), 256, 0, stream>>>(b2a, b2revb, tgt);

    // h0 = h = relu(concat(f_atoms[b2a], f_bonds) @ W_i^T)
    k_build_ai<<<nblk((size_t)kBonds * kKi), 256, 0, stream>>>(f_atoms, f_bonds, b2a, Abuf);
    gemm_wmma_bf16<0><<<nblk((size_t)MT_B * kNGrp * 32), 256, 0, stream>>>(
        Abuf, kKi, WiP, kKi, MT_B, kKi / 32, nullptr, h, h0, nullptr, nullptr, nullptr);

    // Message passing: h = relu(h0 + (atom_msg[b2a] - h[b2revb]) @ W_h^T)
    for (int d = 0; d < kDepth; ++d) {
        k_zero_f32<<<nblk((size_t)kAtoms * kHid), 256, 0, stream>>>(am, (size_t)kAtoms * kHid);
        k_scatter<<<nblk((size_t)kBonds * kHid), 256, 0, stream>>>(h, tgt, am);
        k_build_m<<<nblk((size_t)kBonds * kKh), 256, 0, stream>>>(am, h, b2a, b2revb, Abuf);
        gemm_wmma_bf16<1><<<nblk((size_t)MT_B * kNGrp * 32), 256, 0, stream>>>(
            Abuf, kKh, WhP, kKh, MT_B, kKh / 32, h0, h, nullptr, nullptr, nullptr, nullptr);
    }

    // Final aggregation + fused readout pooling into mol_vecs.
    k_zero_f32<<<nblk((size_t)kAtoms * kHid), 256, 0, stream>>>(am, (size_t)kAtoms * kHid);
    k_scatter<<<nblk((size_t)kBonds * kHid), 256, 0, stream>>>(h, tgt, am);
    k_build_ao<<<nblk((size_t)kAtoms * kKo), 256, 0, stream>>>(f_atoms, am, Abuf);
    k_zero_f32<<<nblk((size_t)kMols * kHid), 256, 0, stream>>>(mv, (size_t)kMols * kHid);
    gemm_wmma_bf16<2><<<nblk((size_t)MT_A * kNGrp * 32), 256, 0, stream>>>(
        Abuf, kKo, WoP, kKo, MT_A, kKo / 32, nullptr, nullptr, nullptr, b_o, mol_ids, mv);

    // Tiny fp32 readout GEMMs.
    k_r1<<<nblk((size_t)kMols * kHid), 256, 0, stream>>>(mv, W_r1, b_r1, r1o);
    k_r2<<<nblk((size_t)kMols * kTasks), 256, 0, stream>>>(r1o, W_r2, b_r2, logits);
}
